// CKNLayer_80041010528306
// MI455X (gfx1250) — matrix operand order repaired and checked by reading, not compiled, not run on hardware
//
#include <hip/hip_runtime.h>
#include <hip/hip_bf16.h>

typedef __attribute__((ext_vector_type(16))) _Float16 v16h;
typedef __attribute__((ext_vector_type(8)))  _Float16 v8h;
typedef __attribute__((ext_vector_type(8)))  float    v8f;

#define C_IN   64
#define O_OUT  256
#define HW     64
#define KDIM   576        // C_IN * 9, reordered as (kh*3+kw)*64 + c
#define TROWS  4          // output rows per workgroup
#define NBLK   64         // output channels per workgroup

// LDS partition (bytes)
#define XT_HALFS   (6*66*64)          // halo tile [dy][xx][c], f16 (50688 B)
#define WL_HALFS   (NBLK*KDIM)        // weight block [n][k], f16  (73728 B)
#define SM_XT      0
#define SM_WL      (XT_HALFS*2)
#define SM_PN      (XT_HALFS*2 + WL_HALFS*2)
#define SM_TOTAL   (SM_PN + 256*4)
// pooled-output staging reuses the xt region after it is dead (16384 B << 50688 B)

// ---------------------------------------------------------------------------
// Kernel 1: L2-normalize weight rows, reorder K to (kh,kw,c), convert to f16.
// grid = 256 blocks (one per output channel), 64 threads.
// ---------------------------------------------------------------------------
__global__ void ckn_wprep(const float* __restrict__ w, _Float16* __restrict__ w2) {
    __shared__ float red[64];
    const int o = blockIdx.x;
    const int t = threadIdx.x;
    float s = 0.f;
    for (int k = t; k < KDIM; k += 64) {
        float v = w[o*KDIM + k];
        s += v*v;
    }
    red[t] = s;
    __syncthreads();
    for (int off = 32; off > 0; off >>= 1) {
        if (t < off) red[t] += red[t + off];
        __syncthreads();
    }
    const float inv = 1.f / fmaxf(sqrtf(red[0]), 1e-12f);
    for (int k = t; k < KDIM; k += 64) {
        const int c  = k / 9;           // torch unfold order: (c, kh, kw)
        const int kk = k % 9;           // kh*3 + kw
        w2[o*KDIM + kk*64 + c] = (_Float16)(w[o*KDIM + k] * inv);
    }
}

// ---------------------------------------------------------------------------
// Kernel 2: implicit-GEMM conv via v_wmma_f32_16x16x32_f16 + RBF + 2x2 pool.
// grid = (4 nblocks, 16 rowblocks, 32 images), 256 threads = 8 waves.
// Wave w: rpair = w&1 (rows 2*rpair, 2*rpair+1), wseg = w>>1 (16-px column seg).
// ---------------------------------------------------------------------------
__global__ __launch_bounds__(256)
void ckn_main(const float* __restrict__ x, const _Float16* __restrict__ w2,
              float* __restrict__ out) {
    extern __shared__ char smem[];
    _Float16* xt = (_Float16*)(smem + SM_XT);  // [6][66][64] f16, halo + zero pad
    _Float16* wl = (_Float16*)(smem + SM_WL);  // [64][576]  f16
    float*    pn = (float*)   (smem + SM_PN);  // [4*64] patch norms

    const int nblock = blockIdx.x;             // 0..3  : output-channel block
    const int rowblk = blockIdx.y;             // 0..15 : 4-row block
    const int b      = blockIdx.z;             // image

    const int tid   = threadIdx.x;
    const int wave  = tid >> 5;
    const int lane  = tid & 31;
    const int r     = lane & 15;
    const int hi    = lane >> 4;
    const int rpair = wave & 1;
    const int wseg  = wave >> 1;

    // ---- stage weight block (64 x 576 f16 = 72KB) into LDS, contiguous ----
    {
        const uint4* src = (const uint4*)(w2 + (size_t)nblock * NBLK * KDIM);
        uint4*       dst = (uint4*)wl;
        for (int i = tid; i < (NBLK * KDIM) / 8; i += 256) dst[i] = src[i];
    }

    // ---- stage x halo tile into LDS as f16, pixel-major [dy][xx][c] ----
    for (int p = tid; p < 6 * 66; p += 256) {
        const int dy = p / 66;
        const int xx = p % 66;
        const int gy = rowblk * TROWS - 1 + dy;
        const int gx = xx - 1;
        const bool ok = (gy >= 0) & (gy < HW) & (gx >= 0) & (gx < HW);
        for (int c8 = 0; c8 < 8; ++c8) {
            v8h h;
            if (ok) {
                const float* px = x + (((size_t)b * C_IN + c8 * 8) * HW + gy) * HW + gx;
                #pragma unroll
                for (int i = 0; i < 8; ++i) h[i] = (_Float16)px[(size_t)i * HW * HW];
            } else {
                #pragma unroll
                for (int i = 0; i < 8; ++i) h[i] = (_Float16)0.f;
            }
            *(v8h*)(xt + p * 64 + c8 * 8) = h;
        }
    }
    __syncthreads();

    // ---- main GEMM: K = 9 taps x 2 halves of 32 channels ----
    v8f acc[2][4];
    #pragma unroll
    for (int m = 0; m < 2; ++m)
        #pragma unroll
        for (int n = 0; n < 4; ++n)
            acc[m][n] = (v8f){0.f,0.f,0.f,0.f,0.f,0.f,0.f,0.f};

    for (int kpos = 0; kpos < 9; ++kpos) {
        const int kh = kpos / 3, kw = kpos % 3;
        #pragma unroll
        for (int cc = 0; cc < 2; ++cc) {
            const int c0 = cc * 32;
            // A fragments: 16 contiguous K at hi*16 (matches B)
            v16h af[2];
            #pragma unroll
            for (int m = 0; m < 2; ++m) {
                const int row  = rpair * 2 + m;
                const int dy   = row + kh;
                const int xx   = wseg * 16 + r + kw;
                const int base = (dy * 66 + xx) * 64 + c0 + hi * 16;
                const v8h lo = *(const v8h*)(xt + base);
                const v8h hh = *(const v8h*)(xt + base + 8);
                #pragma unroll
                for (int i = 0; i < 8; ++i) { af[m][i] = lo[i]; af[m][i + 8] = hh[i]; }
            }
            // B fragments: lane = N column, 16 contiguous K at hi*16
            v16h bf[4];
            #pragma unroll
            for (int n = 0; n < 4; ++n) {
                const int ol   = n * 16 + r;
                const int base = ol * KDIM + kpos * 64 + c0 + hi * 16;
                bf[n] = *(const v16h*)(wl + base);
            }
            #pragma unroll
            for (int m = 0; m < 2; ++m)
                #pragma unroll
                for (int n = 0; n < 4; ++n)
                    acc[m][n] = __builtin_amdgcn_wmma_f32_16x16x32_f16(
                        false, af[m], false, bf[n], (short)0, acc[m][n], false, false);
        }
    }

    // ---- patch L2 norms: one thread per tile pixel (4 rows x 64 cols) ----
    {
        const int prow = tid >> 6, pcol = tid & 63;
        float s = 0.f;
        for (int dy = 0; dy < 3; ++dy)
            for (int dx = 0; dx < 3; ++dx) {
                const int base = ((prow + dy) * 66 + (pcol + dx)) * 64;
                for (int c8 = 0; c8 < 8; ++c8) {
                    const v8h h = *(const v8h*)(xt + base + c8 * 8);
                    #pragma unroll
                    for (int i = 0; i < 8; ++i) { const float f = (float)h[i]; s += f * f; }
                }
            }
        pn[tid] = sqrtf(fmaxf(s, 0.f)) + 1e-8f;
    }
    __syncthreads();   // pnorm ready; xt is dead from here on

    // ---- RBF epilogue + 2x2 avg pool -> LDS staging (reuse xt region) ----
    float* pool = (float*)smem;   // [64 ch][2 prow][32 pcol] f32 = 16KB

    float invp[2][8];
    #pragma unroll
    for (int m = 0; m < 2; ++m) {
        const int row = rpair * 2 + m;
        #pragma unroll
        for (int j = 0; j < 8; ++j)
            invp[m][j] = 1.f / pn[row * 64 + wseg * 16 + hi * 8 + j];
    }

    #pragma unroll
    for (int n = 0; n < 4; ++n) {
        float e0[8], e1[8];
        #pragma unroll
        for (int j = 0; j < 8; ++j) {
            e0[j] = __expf(acc[0][n][j] * invp[0][j] - 1.f);
            e1[j] = __expf(acc[1][n][j] * invp[1][j] - 1.f);
        }
        const int ol = n * 16 + r;                 // C layout: N = lane&15
        #pragma unroll
        for (int pr = 0; pr < 4; ++pr) {
            const float v  = 0.25f * (e0[2*pr] + e0[2*pr+1] + e1[2*pr] + e1[2*pr+1]);
            const int   pc = wseg * 8 + hi * 4 + pr;
            pool[ol * 64 + rpair * 32 + pc] = v;
        }
    }
    __syncthreads();

    // ---- coalesced 128-bit stores: 64 consecutive floats per channel ----
    // out[((b*256 + o)*32 + ph)*32 + pw]; per channel o the (prow,pcol) pair is
    // 64 contiguous floats starting at base + ol*1024.
    const size_t base = ((size_t)(b * O_OUT + nblock * NBLK)) * 1024 + (size_t)rowblk * 64;
    const float4* poolv = (const float4*)pool;
    for (int i4 = tid; i4 < 1024; i4 += 256) {
        const int ol  = i4 >> 4;
        const int rem = i4 & 15;
        *(float4*)(out + base + (size_t)ol * 1024 + rem * 4) = poolv[i4];
    }
}

// ---------------------------------------------------------------------------
extern "C" void kernel_launch(void* const* d_in, const int* in_sizes, int n_in,
                              void* d_out, int out_size, void* d_ws, size_t ws_size,
                              hipStream_t stream) {
    const float* x = (const float*)d_in[0];   // [32,64,64,64] f32
    const float* w = (const float*)d_in[1];   // [256,576] f32
    float* out = (float*)d_out;               // [32,256,32,32] f32
    _Float16* w2 = (_Float16*)d_ws;           // 576*256 f16 = 288KB scratch

    ckn_wprep<<<O_OUT, 64, 0, stream>>>(w, w2);

    (void)hipFuncSetAttribute((const void*)ckn_main,
                              hipFuncAttributeMaxDynamicSharedMemorySize, SM_TOTAL);
    dim3 grid(4, 16, 32);
    ckn_main<<<grid, 256, SM_TOTAL, stream>>>(x, w2, out);
}
